// GNNBase_48481590837648
// MI455X (gfx1250) — matrix operand order repaired and checked by reading, not compiled, not run on hardware
//
#include <hip/hip_runtime.h>

typedef _Float16 f16;
typedef __attribute__((ext_vector_type(16))) _Float16 v16h;
typedef __attribute__((ext_vector_type(8)))  _Float16 v8h;
typedef __attribute__((ext_vector_type(2)))  __fp16   v2fp16;  // cvt_pkrtz result
typedef __attribute__((ext_vector_type(8)))  float    v8f;

#define HID   256      // hidden width
#define TILE_E 64      // edges per workgroup
#define MT     4       // M-tiles (16 rows each) per wave
#define NWAVE  8       // 256 threads, wave32
#define LDH   296      // padded LDS row stride (halves): cols 0..255 data,
                       // col 256 = 1.0 (bias column), 257..287 zero, + pad
#define LD1   104      // padded LDS row stride for 96-wide layer-1 input
#define K1P   96       // layer-1 K padded (68 data + bias col 68 -> 96)
#define KC1   3        // K-chunks of 32 for layer 1
#define KC2   9        // K-chunks of 32 for layers 2/3 (256 data + bias col)

// 16-lane butterfly sum: exactly 4 VALU instructions via v_add_f32_dpp
// (dst = dpp(src0) + src1). xor1, xor2, xor7, xor15 within each DPP row.
// All lanes must be active (true here: no divergence before use).
__device__ __forceinline__ float row16_sum(float v) {
  float x = v;
  asm("v_add_f32_dpp %0, %0, %0 quad_perm:[1,0,3,2] row_mask:0xf bank_mask:0xf bound_ctrl:1\n\t"
      "v_add_f32_dpp %0, %0, %0 quad_perm:[2,3,0,1] row_mask:0xf bank_mask:0xf bound_ctrl:1\n\t"
      "v_add_f32_dpp %0, %0, %0 row_half_mirror row_mask:0xf bank_mask:0xf bound_ctrl:1\n\t"
      "v_add_f32_dpp %0, %0, %0 row_mirror row_mask:0xf bank_mask:0xf bound_ctrl:1"
      : "+v"(x));
  return x;   // all 16 lanes of the row-group now hold the total
}

// ---------------------------------------------------------------------------
// Weight pre-pack: f32 row-major W[k][n] -> f16 WMMA B-fragment layout.
//  - previous layer's LayerNorm gamma folded into rows (W' = g_prev ⊙ W)
//  - bias (pre-beta-folded for layers 2/3) appended as row k == Kreal, matched
//    by a constant-1 activation column, so the GEMM accumulates the bias.
//  - permK: the producing layer stores its output K-interleaved so each lane
//    writes its two values adjacently (kpos = w*32 + 2m + t holds logical
//    column w*32 + t*16 + m); the pack reads rows through that permutation.
// B operand of V_WMMA_F32_16X16X32_F16 is 32(K) x 16(N), 8 VGPRs per lane:
//   lane L holds column N = L%16, K = (L/16)*8 + (h&7) + (h>>3)*16  (h = half index 0..15)
// ---------------------------------------------------------------------------
__global__ void pack_w_kernel(const float* __restrict__ W,
                              const float* __restrict__ gprev,   // may be null
                              const float* __restrict__ biasRow, // row k==Kreal
                              f16* __restrict__ out, int Kreal, int KC,
                              int permK) {
  int idx   = blockIdx.x * 256 + threadIdx.x;
  int total = 16 * KC * 32 * 16;           // ntiles * kchunks * lanes * halves
  if (idx >= total) return;
  int h    = idx & 15;
  int lane = (idx >> 4) & 31;
  int kc   = (idx >> 9) % KC;
  int nt   = idx / (KC * 512);
  int n = nt * 16 + (lane & 15);
  int k = kc * 32 + ((lane >> 4) << 3) + (h & 7) + ((h >> 3) << 4);   // packed K position
  float v = 0.0f;
  if (k < Kreal) {
    int kl = k;
    if (permK) kl = (k & ~31) | (((k >> 1) & 15) + ((k & 1) << 4));
    v = W[(size_t)kl * HID + n];
    if (gprev) v *= gprev[kl];
  } else if (k == Kreal) {
    v = biasRow[n];
  }
  out[idx] = (f16)v;
}

// Fold previous layer's beta into this layer's bias: b'[n] = b[n] + sum_k beta[k]*W[k][n]
__global__ void fold_bias_kernel(const float* __restrict__ b,
                                 const float* __restrict__ beta_prev,
                                 const float* __restrict__ W,
                                 float* __restrict__ out) {
  const int n = threadIdx.x;   // 256 threads
  float acc = b[n];
  for (int k = 0; k < HID; ++k) acc = fmaf(beta_prev[k], W[(size_t)k * HID + n], acc);
  out[n] = acc;
}

// ---------------------------------------------------------------------------
// One fused Linear(+bias via ones-column) -> ReLU -> LayerNorm block on a
// 64-edge tile. gamma/beta of this layer are folded into the NEXT layer's
// weights/bias, so intermediate output is z = h*rstd + (-mu*rstd), written
// K-interleaved as packed half2 (one cvt_pk + one b32 store per pair).
// Final layer applies gamma3/beta3 explicitly and scatter-adds to global out.
// ---------------------------------------------------------------------------
__device__ __forceinline__ void mlp_layer(
    const f16* __restrict__ A, int lda, int kcCount,
    const f16* __restrict__ Wp,
    f16* __restrict__ Bout,                  // next-layer LDS buffer (stride LDH)
    float* __restrict__ outG,                // global output (final layer)
    const int* __restrict__ dstIdx,          // LDS: destination node per row
    bool finalLayer,
    const float* __restrict__ gam3, const float* __restrict__ bet3,
    float* __restrict__ redS, float* __restrict__ redQ,  // LDS [NWAVE*TILE_E]
    float* __restrict__ scl, float* __restrict__ sft)    // LDS [TILE_E]
{
  const int tid  = threadIdx.x;
  const int wave = tid >> 5;
  const int lane = tid & 31;
  const int m    = lane & 15;   // N-column within tile / M-row within A frag
  const int g    = lane >> 4;   // half-group

  const v8f zero = {0.f, 0.f, 0.f, 0.f, 0.f, 0.f, 0.f, 0.f};
  v8f c[MT][2];                 // [mtile][ntile]
#pragma unroll
  for (int mt = 0; mt < MT; ++mt) { c[mt][0] = zero; c[mt][1] = zero; }

  union AV { v16h v; v8h h[2]; };

  for (int kc = 0; kc < kcCount; ++kc) {
    const int k0 = kc * 32 + g * 8;
    // B fragments for this wave's two N-tiles (reused across all 4 M-tiles).
    const int nt0 = wave * 2;
    const v16h b0 = *(const v16h*)(Wp + (((size_t)nt0       * kcCount + kc) * 32 + lane) * 16);
    const v16h b1 = *(const v16h*)(Wp + (((size_t)(nt0 + 1) * kcCount + kc) * 32 + lane) * 16);
#pragma unroll
    for (int mt = 0; mt < MT; ++mt) {
      // A fragment: 16-bit A 16x32 layout -> two contiguous 8-half runs:
      // k in [g*8, g*8+8) and [16+g*8, 16+g*8+8) of this K-chunk.
      AV a;
      const f16* r0 = A + (size_t)(mt * 16 + m) * lda + k0;
      a.h[0] = *(const v8h*)(r0);
      a.h[1] = *(const v8h*)(r0 + 16);
      c[mt][0] = __builtin_amdgcn_wmma_f32_16x16x32_f16(
          false, a.v, false, b0, (short)0, c[mt][0], false, false);
      c[mt][1] = __builtin_amdgcn_wmma_f32_16x16x32_f16(
          false, a.v, false, b1, (short)0, c[mt][1], false, false);
    }
  }

  // ReLU (bias already accumulated via the ones-column GEMM row).
#pragma unroll
  for (int mt = 0; mt < MT; ++mt)
#pragma unroll
    for (int r = 0; r < 8; ++r) {
      c[mt][0][r] = fmaxf(c[mt][0][r], 0.f);
      c[mt][1][r] = fmaxf(c[mt][1][r], 0.f);
    }

  // Row-wise partial sums over this wave's 32 columns via DPP butterflies.
  // C layout: half-group g covers rows mt*16 + g*8 + r.
#pragma unroll
  for (int mt = 0; mt < MT; ++mt)
#pragma unroll
    for (int r = 0; r < 8; ++r) {
      float sv = c[mt][0][r] + c[mt][1][r];
      float qv = fmaf(c[mt][0][r], c[mt][0][r], c[mt][1][r] * c[mt][1][r]);
      sv = row16_sum(sv);
      qv = row16_sum(qv);
      if (m == 0) {
        const int row = mt * 16 + g * 8 + r;
        redS[wave * TILE_E + row] = sv;
        redQ[wave * TILE_E + row] = qv;
      }
    }
  __syncthreads();

  // Cross-wave combine: one thread per row computes scale = rstd, shift = -mu*rstd.
  if (tid < TILE_E) {
    float ts = 0.f, tq = 0.f;
#pragma unroll
    for (int w = 0; w < NWAVE; ++w) {
      ts += redS[w * TILE_E + tid];
      tq += redQ[w * TILE_E + tid];
    }
    const float mean = ts * (1.0f / (float)HID);
    const float var  = tq * (1.0f / (float)HID) - mean * mean;
    const float rstd = rsqrtf(var + 1e-5f);
    scl[tid] = rstd;
    sft[tid] = -mean * rstd;
  }
  __syncthreads();

  if (finalLayer) {
    const int n0 = wave * 32 + m;
    const int n1 = n0 + 16;
    const float ga0 = gam3[n0], ga1 = gam3[n1];
    const float be0 = bet3[n0], be1 = bet3[n1];
#pragma unroll
    for (int mt = 0; mt < MT; ++mt)
#pragma unroll
      for (int r = 0; r < 8; ++r) {
        const int row = mt * 16 + g * 8 + r;
        const float z0 = fmaf(c[mt][0][r], scl[row], sft[row]);
        const float z1 = fmaf(c[mt][1][r], scl[row], sft[row]);
        const int d = dstIdx[row];
        if (d >= 0) {
          atomicAdd(outG + (size_t)d * HID + n0, fmaf(z0, ga0, be0));
          atomicAdd(outG + (size_t)d * HID + n1, fmaf(z1, ga1, be1));
        }
      }
  } else {
    // K-interleaved packed store: lane's two outputs land adjacently.
    const int kbase = wave * 32 + m * 2;
#pragma unroll
    for (int mt = 0; mt < MT; ++mt)
#pragma unroll
      for (int r = 0; r < 8; ++r) {
        const int row = mt * 16 + g * 8 + r;
        const float sc = scl[row], sh = sft[row];
        const v2fp16 pk = __builtin_amdgcn_cvt_pkrtz(
            fmaf(c[mt][0][r], sc, sh), fmaf(c[mt][1][r], sc, sh));
        *(v2fp16*)(Bout + (size_t)row * LDH + kbase) = pk;
      }
    // Bias/ones + zero pad columns 256..287 for the next layer's K-chunk 8.
    if (tid < TILE_E) {
      const v8h onep = {(f16)1.0f, (f16)0.f, (f16)0.f, (f16)0.f,
                        (f16)0.f,  (f16)0.f, (f16)0.f, (f16)0.f};
      const v8h zp   = {(f16)0.f, (f16)0.f, (f16)0.f, (f16)0.f,
                        (f16)0.f, (f16)0.f, (f16)0.f, (f16)0.f};
      f16* p = Bout + (size_t)tid * LDH + 256;
      *(v8h*)(p)      = onep;
      *(v8h*)(p + 8)  = zp;
      *(v8h*)(p + 16) = zp;
      *(v8h*)(p + 24) = zp;
    }
  }
  __syncthreads();   // A-buffer fully consumed; Bout complete for next layer
}

// ---------------------------------------------------------------------------
// Fused per-edge MLP + scatter-add.
// ---------------------------------------------------------------------------
__global__ __launch_bounds__(256) void gnn_fused_kernel(
    const float* __restrict__ node_feat, const int* __restrict__ node_type,
    const int* __restrict__ edge_index, const float* __restrict__ edge_attr,
    const float* __restrict__ embed,
    const f16* __restrict__ W1p, const f16* __restrict__ W2p,
    const f16* __restrict__ W3p,
    const float* __restrict__ g3, const float* __restrict__ be3,
    float* __restrict__ out, int E)
{
  __shared__ f16   bufA[TILE_E * LDH];
  __shared__ f16   bufB[TILE_E * LDH];
  __shared__ float redS[NWAVE * TILE_E];
  __shared__ float redQ[NWAVE * TILE_E];
  __shared__ float scl[TILE_E];
  __shared__ float sft[TILE_E];
  __shared__ int   dstIdx[TILE_E];

  const int  tid = threadIdx.x;
  const long eb  = (long)blockIdx.x * TILE_E;

  if (tid < TILE_E) {
    const long e = eb + tid;
    dstIdx[tid] = (e < E) ? edge_index[(size_t)E + e] : -1;
  }

  // Gather layer-1 input tile: 4 threads per row, float4 loads, 16B LDS stores.
  // Row layout: [feat_j(0..31) | emb_j(32..63) | edge_attr(64..67) | 1 | 0-pad]
  {
    const int  r = tid >> 2;          // row 0..63
    const int  q = tid & 3;           // quarter 0..3 (8 cols each segment)
    const long e = eb + r;
    const bool valid = (e < E);
    const int  s = valid ? edge_index[e] : 0;

    float4 f0 = {0.f,0.f,0.f,0.f}, f1 = {0.f,0.f,0.f,0.f};
    float4 m0 = {0.f,0.f,0.f,0.f}, m1 = {0.f,0.f,0.f,0.f};
    if (valid) {
      const float4* fp = (const float4*)(node_feat + (size_t)s * 32 + q * 8);
      f0 = fp[0]; f1 = fp[1];
      const int ty = node_type[s];
      const float4* ep = (const float4*)(embed + (size_t)ty * 32 + q * 8);
      m0 = ep[0]; m1 = ep[1];
    }
    v8h hf, hm;
    hf[0]=(f16)f0.x; hf[1]=(f16)f0.y; hf[2]=(f16)f0.z; hf[3]=(f16)f0.w;
    hf[4]=(f16)f1.x; hf[5]=(f16)f1.y; hf[6]=(f16)f1.z; hf[7]=(f16)f1.w;
    hm[0]=(f16)m0.x; hm[1]=(f16)m0.y; hm[2]=(f16)m0.z; hm[3]=(f16)m0.w;
    hm[4]=(f16)m1.x; hm[5]=(f16)m1.y; hm[6]=(f16)m1.z; hm[7]=(f16)m1.w;
    f16* rowp = bufA + (size_t)r * LD1;
    *(v8h*)(rowp + q * 8)      = hf;
    *(v8h*)(rowp + 32 + q * 8) = hm;

    if (q == 0) {
      float4 a = {0.f,0.f,0.f,0.f};
      if (valid) a = *(const float4*)(edge_attr + (size_t)e * 4);
      v8h t0;
      t0[0]=(f16)a.x; t0[1]=(f16)a.y; t0[2]=(f16)a.z; t0[3]=(f16)a.w;
      t0[4]= valid ? (f16)1.0f : (f16)0.f;           // bias column (W1 row 68)
      t0[5]=(f16)0.f; t0[6]=(f16)0.f; t0[7]=(f16)0.f;
      const v8h zp = {(f16)0.f,(f16)0.f,(f16)0.f,(f16)0.f,
                      (f16)0.f,(f16)0.f,(f16)0.f,(f16)0.f};
      *(v8h*)(rowp + 64) = t0;
      *(v8h*)(rowp + 72) = zp;
      *(v8h*)(rowp + 80) = zp;
      *(v8h*)(rowp + 88) = zp;
    }
  }
  __syncthreads();

  mlp_layer(bufA, LD1, KC1, W1p, bufB, nullptr, dstIdx, false,
            nullptr, nullptr, redS, redQ, scl, sft);
  mlp_layer(bufB, LDH, KC2, W2p, bufA, nullptr, dstIdx, false,
            nullptr, nullptr, redS, redQ, scl, sft);
  mlp_layer(bufA, LDH, KC2, W3p, nullptr, out, dstIdx, true,
            g3, be3, redS, redQ, scl, sft);
}

extern "C" void kernel_launch(void* const* d_in, const int* in_sizes, int n_in,
                              void* d_out, int out_size, void* d_ws, size_t ws_size,
                              hipStream_t stream) {
  const float* node_feat = (const float*)d_in[0];
  const int*   node_type = (const int*)  d_in[1];
  const int*   edge_idx  = (const int*)  d_in[2];
  const float* edge_attr = (const float*)d_in[3];
  const float* embed     = (const float*)d_in[4];
  const float* W1  = (const float*)d_in[5];
  const float* b1  = (const float*)d_in[6];
  const float* g1  = (const float*)d_in[7];
  const float* be1 = (const float*)d_in[8];
  const float* W2  = (const float*)d_in[9];
  const float* b2  = (const float*)d_in[10];
  const float* g2  = (const float*)d_in[11];
  const float* be2 = (const float*)d_in[12];
  const float* W3  = (const float*)d_in[13];
  const float* b3  = (const float*)d_in[14];
  const float* g3  = (const float*)d_in[15];
  const float* be3 = (const float*)d_in[16];

  const int E = in_sizes[3] / 4;          // edge_attr is [E,4]

  // Workspace: pre-packed f16 weight fragments (~345 KB) + folded biases.
  f16* W1p = (f16*)d_ws;                          // 16*KC1*512 halves
  f16* W2p = W1p + (size_t)16 * KC1 * 512;
  f16* W3p = W2p + (size_t)16 * KC2 * 512;
  float* b2f = (float*)(W3p + (size_t)16 * KC2 * 512);
  float* b3f = b2f + HID;

  (void)hipMemsetAsync(d_out, 0, (size_t)out_size * sizeof(float), stream);

  // Fold beta(l) into b(l+1) first; pack consumes the folded biases as the
  // K==Kreal bias row (matched by the ones-column in the activations).
  fold_bias_kernel<<<1, 256, 0, stream>>>(b2, be1, W2, b2f);
  fold_bias_kernel<<<1, 256, 0, stream>>>(b3, be2, W3, b3f);

  const int t1 = 16 * KC1 * 512;
  const int t2 = 16 * KC2 * 512;
  // W1: K comes from the gather layout (no perm). W2/W3: K-interleaved perm.
  pack_w_kernel<<<(t1 + 255) / 256, 256, 0, stream>>>(W1, nullptr, b1,  W1p, 68,  KC1, 0);
  pack_w_kernel<<<(t2 + 255) / 256, 256, 0, stream>>>(W2, g1,      b2f, W2p, HID, KC2, 1);
  pack_w_kernel<<<(t2 + 255) / 256, 256, 0, stream>>>(W3, g2,      b3f, W3p, HID, KC2, 1);

  const int nblk = (E + TILE_E - 1) / TILE_E;
  gnn_fused_kernel<<<nblk, 256, 0, stream>>>(
      node_feat, node_type, edge_idx, edge_attr, embed,
      W1p, W2p, W3p, g3, be3,
      (float*)d_out, E);
}
